// HGNN_layer_4870492913805
// MI455X (gfx1250) — matrix operand order repaired and checked by reading, not compiled, not run on hardware
//
#include <hip/hip_runtime.h>
#include <hip/hip_bf16.h>

typedef __attribute__((ext_vector_type(16))) _Float16 v16h;
typedef __attribute__((ext_vector_type(2)))  _Float16 v2h;
typedef __attribute__((ext_vector_type(8)))  float    v8f;

// ---------------------------------------------------------------------------
// Pack a 128x128 f32 weight matrix into f16 WMMA B-fragments.
// Layout: frag[kt][nt][lane][i]  (kt: K-tile of 32, nt: N-tile of 16)
//   element i of lane L = W[kt*32 + (i<8 ? i : i+8) + 8*(L/16), nt*16 + (L%16)]
// so each lane's 16 halfs are contiguous (32B) -> one b128x2 load in the GEMM.
// ---------------------------------------------------------------------------
__global__ void pack_w_f16(const float* __restrict__ W, _Float16* __restrict__ out) {
    int tid  = blockIdx.x * blockDim.x + threadIdx.x;   // 0..16383
    int i    = tid & 15;
    int lane = (tid >> 4) & 31;
    int nt   = (tid >> 9) & 7;
    int kt   = tid >> 12;
    int row  = kt * 32 + (i < 8 ? i : i + 8) + 8 * (lane >> 4);
    int col  = nt * 16 + (lane & 15);
    out[tid] = (_Float16)W[row * 128 + col];
}

// ---------------------------------------------------------------------------
// GEMM1: h[16-row tile] = x(f32 -> f16) @ W1p, output f16.
// Block = 256 threads = 8 waves; wave w computes N-tile w (cols 16w..16w+15).
// A-fragment layout (16-bit A 16x32): lane L holds row M=L%16,
//   K = (i<8 ? i : i+8) + 8*(L/16)  for element i.
// ---------------------------------------------------------------------------
__global__ void gemm_x_w1(const float* __restrict__ x,
                          const _Float16* __restrict__ wp,
                          _Float16* __restrict__ out) {
    const int lane = threadIdx.x & 31;
    const int wv   = threadIdx.x >> 5;          // 0..7 -> N tile
    const int m    = lane & 15;
    const int hl   = lane >> 4;                 // half-lane
    const int row  = blockIdx.x * 16 + m;
    const float* xr = x + (long)row * 128;

    v8f acc = {};
#pragma unroll
    for (int kt = 0; kt < 4; ++kt) {
        v16h a;
        const float* xk = xr + kt * 32 + hl * 8;
#pragma unroll
        for (int i = 0; i < 8; ++i)  a[i] = (_Float16)xk[i];
#pragma unroll
        for (int i = 8; i < 16; ++i) a[i] = (_Float16)xk[i + 8];
        v16h b = *(const v16h*)(wp + (((kt * 8 + wv) * 32 + lane) << 4));
        acc = __builtin_amdgcn_wmma_f32_16x16x32_f16(
            false, a, false, b, (short)0, acc, false, false);
    }
    // D layout: element i -> row M = i + 8*hl, col N = m
    _Float16* o = out + ((long)(blockIdx.x * 16 + hl * 8)) * 128 + wv * 16 + m;
#pragma unroll
    for (int i = 0; i < 8; ++i) o[(long)i * 128] = (_Float16)acc[i];
}

// ---------------------------------------------------------------------------
// GEMM2: e1 = edge(f16) @ W2p, output f16. Same structure, A already f16.
// ---------------------------------------------------------------------------
__global__ void gemm_edge_w2(const _Float16* __restrict__ A,
                             const _Float16* __restrict__ wp,
                             _Float16* __restrict__ out) {
    const int lane = threadIdx.x & 31;
    const int wv   = threadIdx.x >> 5;
    const int m    = lane & 15;
    const int hl   = lane >> 4;
    const int row  = blockIdx.x * 16 + m;
    const _Float16* ar = A + (long)row * 128;

    v8f acc = {};
#pragma unroll
    for (int kt = 0; kt < 4; ++kt) {
        v16h a;
        const _Float16* ak = ar + kt * 32 + hl * 8;
#pragma unroll
        for (int i = 0; i < 8; ++i)  a[i] = ak[i];
#pragma unroll
        for (int i = 8; i < 16; ++i) a[i] = ak[i + 8];
        v16h b = *(const v16h*)(wp + (((kt * 8 + wv) * 32 + lane) << 4));
        acc = __builtin_amdgcn_wmma_f32_16x16x32_f16(
            false, a, false, b, (short)0, acc, false, false);
    }
    _Float16* o = out + ((long)(blockIdx.x * 16 + hl * 8)) * 128 + wv * 16 + m;
#pragma unroll
    for (int i = 0; i < 8; ++i) o[(long)i * 128] = (_Float16)acc[i];
}

// ---------------------------------------------------------------------------
// Masked-mean gather over 32 rows of a f16 [*,128] table.
// weight = 1/k over slots with idx>0 (k = popcount); if k==0 -> uniform 1/32
// (softmax of {1,-9e15} underflows masked slots to exactly 0).
// Block = 64 threads, each owns a half2 channel pair. ReLU + f16 store (edge).
// ---------------------------------------------------------------------------
__global__ void agg_edges(const _Float16* __restrict__ rows,
                          const int* __restrict__ idxMat,
                          _Float16* __restrict__ out) {
    __shared__ int sIdx[32];
    const int e = blockIdx.x;
    const int t = threadIdx.x;                  // 0..63
    if (t < 32) sIdx[t] = idxMat[(long)e * 32 + t];
    __syncthreads();

    int k = 0;
#pragma unroll
    for (int n = 0; n < 32; ++n) k += (sIdx[n] > 0) ? 1 : 0;
    const bool all  = (k == 0);
    const float scl = 1.0f / (float)(all ? 32 : k);

    float s0 = 0.f, s1 = 0.f;
#pragma unroll
    for (int n = 0; n < 32; ++n) {
        int id = sIdx[n];
        if (id > 0 || all) {
            v2h v = *(const v2h*)(rows + (long)id * 128 + t * 2);
            s0 += (float)v.x;
            s1 += (float)v.y;
        }
    }
    v2h o;
    o.x = (_Float16)fmaxf(s0 * scl, 0.0f);
    o.y = (_Float16)fmaxf(s1 * scl, 0.0f);
    *(v2h*)(out + (long)e * 128 + t * 2) = o;
}

// Same gather/mean, no ReLU, f32 output (final node features).
__global__ void agg_nodes(const _Float16* __restrict__ rows,
                          const int* __restrict__ idxMat,
                          float* __restrict__ out) {
    __shared__ int sIdx[32];
    const int u = blockIdx.x;
    const int t = threadIdx.x;
    if (t < 32) sIdx[t] = idxMat[(long)u * 32 + t];
    __syncthreads();

    int k = 0;
#pragma unroll
    for (int n = 0; n < 32; ++n) k += (sIdx[n] > 0) ? 1 : 0;
    const bool all  = (k == 0);
    const float scl = 1.0f / (float)(all ? 32 : k);

    float s0 = 0.f, s1 = 0.f;
#pragma unroll
    for (int n = 0; n < 32; ++n) {
        int id = sIdx[n];
        if (id > 0 || all) {
            v2h v = *(const v2h*)(rows + (long)id * 128 + t * 2);
            s0 += (float)v.x;
            s1 += (float)v.y;
        }
    }
    float2 o = make_float2(s0 * scl, s1 * scl);
    *(float2*)(out + (long)u * 128 + t * 2) = o;
}

// ---------------------------------------------------------------------------
// Inputs (setup_inputs order):
//  0 x[N,128] f32 | 1 seq[E,32] i32 | 2 text2emb (unused) | 3 useq[U,32] i32
//  4 data_idx (unused) | 5 weight1[128,128] f32 | 6 weight2[128,128] f32
//  7 weight3 (unused)
// Output: node [U,128] f32.
// ---------------------------------------------------------------------------
extern "C" void kernel_launch(void* const* d_in, const int* in_sizes, int n_in,
                              void* d_out, int out_size, void* d_ws, size_t ws_size,
                              hipStream_t stream) {
    const float* x   = (const float*)d_in[0];
    const int*   seq = (const int*)d_in[1];
    const int*   useq= (const int*)d_in[3];
    const float* w1  = (const float*)d_in[5];
    const float* w2  = (const float*)d_in[6];
    float* out = (float*)d_out;

    const int N = in_sizes[0] / 128;   // 100000
    const int E = in_sizes[1] / 32;    // 50000
    const int U = in_sizes[3] / 32;    // 50000

    char* ws = (char*)d_ws;
    size_t off = 0;
    _Float16* w1p   = (_Float16*)(ws + off); off += 128 * 128 * sizeof(_Float16); // packed B frags
    _Float16* w2p   = (_Float16*)(ws + off); off += 128 * 128 * sizeof(_Float16);
    off = (off + 255) & ~(size_t)255;
    _Float16* hF    = (_Float16*)(ws + off); off += (size_t)N * 128 * sizeof(_Float16);
    off = (off + 255) & ~(size_t)255;
    _Float16* edgeF = (_Float16*)(ws + off); off += (size_t)E * 128 * sizeof(_Float16);
    off = (off + 255) & ~(size_t)255;
    _Float16* e1F   = (_Float16*)(ws + off);

    pack_w_f16<<<64, 256, 0, stream>>>(w1, w1p);
    pack_w_f16<<<64, 256, 0, stream>>>(w2, w2p);

    gemm_x_w1<<<N / 16, 256, 0, stream>>>(x, w1p, hF);          // h = x @ W1 (f16 out)
    agg_edges<<<E, 64, 0, stream>>>(hF, seq, edgeF);            // masked mean + ReLU
    gemm_edge_w2<<<E / 16, 256, 0, stream>>>(edgeF, w2p, e1F);  // e1 = edge @ W2
    agg_nodes<<<U, 64, 0, stream>>>(e1F, useq, out);            // masked mean -> f32 out
}
